// OurModel_53094385713925
// MI455X (gfx1250) — matrix operand order repaired and checked by reading, not compiled, not run on hardware
//
#include <hip/hip_runtime.h>
#include <math.h>

// ---------------------------------------------------------------------------
// Slot-attention + LSTM tagger for MI455X (gfx1250, wave32, WMMA).
//
//  * All dense projections via v_wmma_f32_16x16x32_bf16 (bf16 in, f32 acc).
//  * LSTM input-side gates precomputed for ALL timesteps as parallel GEMMs;
//    only h@W_hh^T (M=128,N=2048,K=512) + elementwise cell stay sequential.
//  * Large GEMMs (M=16384) use 32x64 wave tiles (A-pair shares B loads:
//    1.5 b128 loads per WMMA); latency-bound M=128 GEMMs use 16x64 tiles
//    to keep 256 waves in flight per step.
// ---------------------------------------------------------------------------

typedef __attribute__((ext_vector_type(16))) __bf16 v16bf;
typedef __attribute__((ext_vector_type(8)))  __bf16 v8bf;
typedef __attribute__((ext_vector_type(8)))  float  v8f;

#define B_   128
#define T_   128
#define H_   768
#define HEADS_ 12
#define DK_  64
#define RIN_ 512
#define RH_  512
#define NT_  10

__device__ __forceinline__ float bf2f(__bf16 x) {
  unsigned short h = __builtin_bit_cast(unsigned short, x);
  unsigned u = ((unsigned)h) << 16;
  return __builtin_bit_cast(float, u);
}
__device__ __forceinline__ __bf16 f2bf(float f) {
  unsigned u = __builtin_bit_cast(unsigned, f);
  unsigned r = u + 0x7FFFu + ((u >> 16) & 1u);   // round-to-nearest-even
  unsigned short h = (unsigned short)(r >> 16);
  return __builtin_bit_cast(__bf16, h);
}
__device__ __forceinline__ float sigm(float x) { return 1.0f / (1.0f + expf(-x)); }

// -------------------------- elementwise helpers ----------------------------

__global__ void f32_to_bf16_kernel(const float* __restrict__ in,
                                   __bf16* __restrict__ out, long n) {
  long i = (long)blockIdx.x * blockDim.x + threadIdx.x;
  if (i < n) out[i] = f2bf(in[i]);
}

__global__ void bias_combine_kernel(const float* __restrict__ a,
                                    const float* __restrict__ b,
                                    float* __restrict__ out, int n) {
  int i = blockIdx.x * blockDim.x + threadIdx.x;
  if (i < n) out[i] = a[i] + b[i];
}

__global__ void init_state_kernel(float* __restrict__ c,
                                  __bf16* __restrict__ hbf, int n) {
  int i = blockIdx.x * blockDim.x + threadIdx.x;
  if (i < n) { c[i] = 0.0f; hbf[i] = f2bf(0.0f); }
}

// lstm_in[b,t,r] = sigmoid(pre_x[b*T+t, r] + A_ctx[b, r])  -> bf16
__global__ void lstm_in_kernel(const float* __restrict__ prex,
                               const float* __restrict__ actx,
                               __bf16* __restrict__ out) {
  long i = (long)blockIdx.x * blockDim.x + threadIdx.x;
  const long n = (long)B_ * T_ * RIN_;
  if (i < n) {
    long row = i >> 9;            // /RIN_
    int  r   = (int)(i & (RIN_ - 1));
    int  b   = (int)(row >> 7);   // /T_
    out[i] = f2bf(sigm(prex[i] + actx[b * RIN_ + r]));
  }
}

// -------------------------- WMMA GEMM --------------------------------------
// C[M,N] = A[M,K] @ W[N,K]^T  (+bias[n]) (+addsrc[m*stride+n])
// Wave computes (MT*16, 64) tile; the MT A operands share every B operand.

__device__ __forceinline__ v16bf load_a_tile(const __bf16* __restrict__ base,
                                             int ld, int lane) {
  // A operand (16x32): lane = M row; lanes<16: K 0-7 & 16-23; lanes>=16: 8-15 & 24-31
  int r  = lane & 15;
  int k0 = (lane & 16) ? 8 : 0;
  const __bf16* p = base + (long)r * ld;
  v8bf lo = *(const v8bf*)(p + k0);
  v8bf hi = *(const v8bf*)(p + k0 + 16);
  v16bf out;
#pragma unroll
  for (int i = 0; i < 8; ++i) { out[i] = lo[i]; out[8 + i] = hi[i]; }
  return out;
}

__device__ __forceinline__ v16bf load_b_tile(const __bf16* __restrict__ base,
                                             int ld, int lane) {
  // B operand (32x16): lane = N col; lanes<16: K 0-7 & 16-23; lanes>=16: 8-15 & 24-31
  int n    = lane & 15;
  int koff = (lane & 16) ? 8 : 0;
  const __bf16* p = base + (long)n * ld + koff;
  v8bf lo = *(const v8bf*)(p);
  v8bf hi = *(const v8bf*)(p + 16);
  v16bf out;
#pragma unroll
  for (int i = 0; i < 8; ++i) { out[i] = lo[i]; out[8 + i] = hi[i]; }
  return out;
}

template <int MT>
__global__ void gemm_bf16_kernel(const __bf16* __restrict__ A, int lda,
                                 const __bf16* __restrict__ W, int ldw,
                                 float* __restrict__ Cf,
                                 __bf16* __restrict__ Cbf, int ldc,
                                 const float* __restrict__ bias,
                                 const float* __restrict__ addsrc,
                                 long add_row_stride,
                                 int M, int N, int K) {
  const int lane = threadIdx.x & 31;
  const int wave = threadIdx.x >> 5;
  const long gw  = (long)blockIdx.x * (blockDim.x >> 5) + wave;
  const int  nstr  = N >> 6;
  const long total = (long)(M / (16 * MT)) * nstr;
  if (gw >= total) return;              // wave-uniform: EXEC stays all-ones

  const int m0 = (int)(gw / nstr) * (16 * MT);
  const int n0 = (int)(gw % nstr) << 6;

  v8f acc[MT][4] = {};
  const __bf16* Ap[MT];
#pragma unroll
  for (int i = 0; i < MT; ++i) Ap[i] = A + (long)(m0 + 16 * i) * lda;
  const __bf16* Wp[4] = { W + (long)(n0)      * ldw,
                          W + (long)(n0 + 16) * ldw,
                          W + (long)(n0 + 32) * ldw,
                          W + (long)(n0 + 48) * ldw };

  for (int k = 0; k < K; k += 32) {
    v16bf a[MT];
#pragma unroll
    for (int i = 0; i < MT; ++i) a[i] = load_a_tile(Ap[i] + k, lda, lane);
#pragma unroll
    for (int j = 0; j < 4; ++j) {
      v16bf b = load_b_tile(Wp[j] + k, ldw, lane);
#pragma unroll
      for (int i = 0; i < MT; ++i) {
        acc[i][j] = __builtin_amdgcn_wmma_f32_16x16x32_bf16(
            false, a[i], false, b, (short)0, acc[i][j], false, false);
      }
    }
  }

  const int mloc = (lane >> 4) << 3;    // 0 or 8
  const int nl   = lane & 15;
#pragma unroll
  for (int i = 0; i < MT; ++i) {
#pragma unroll
    for (int j = 0; j < 4; ++j) {
      int n = n0 + (j << 4) + nl;
      float badd = bias ? bias[n] : 0.0f;
#pragma unroll
      for (int v = 0; v < 8; ++v) {
        int m = m0 + 16 * i + mloc + v;
        float val = acc[i][j][v] + badd;
        if (addsrc) val += addsrc[(long)m * add_row_stride + n];
        if (Cf) Cf[(long)m * ldc + n] = val;
        else    Cbf[(long)m * ldc + n] = f2bf(val);
      }
    }
  }
}

// -------------------------- attention (tiny) -------------------------------
// One block per (b, head). 128 threads: thread t owns score t.

__global__ void attn_kernel(const float* __restrict__ Q,
                            const __bf16* __restrict__ Kb,
                            const __bf16* __restrict__ Vb,
                            float* __restrict__ Xout) {
  int bh = blockIdx.x;
  int b = bh / HEADS_, h = bh % HEADS_;
  int t = threadIdx.x;
  __shared__ float sc[T_];
  __shared__ float red[T_];

  const float*  q    = Q + (long)b * H_ + h * DK_;
  const __bf16* krow = Kb + ((long)(b * T_ + t)) * H_ + h * DK_;
  float acc = 0.0f;
#pragma unroll 8
  for (int d = 0; d < DK_; ++d) acc += q[d] * bf2f(krow[d]);
  acc *= 0.125f;                                   // 1/sqrt(64)
  sc[t] = acc; red[t] = acc;
  __syncthreads();
  for (int off = 64; off; off >>= 1) {
    if (t < off) red[t] = fmaxf(red[t], red[t + off]);
    __syncthreads();
  }
  float mx = red[0];
  __syncthreads();
  float e = expf(sc[t] - mx);
  sc[t] = e; red[t] = e;
  __syncthreads();
  for (int off = 64; off; off >>= 1) {
    if (t < off) red[t] += red[t + off];
    __syncthreads();
  }
  float inv = 1.0f / red[0];
  if (t < DK_) {
    const __bf16* vb = Vb + (long)b * T_ * H_ + h * DK_ + t;
    float o = 0.0f;
    for (int k = 0; k < T_; ++k) o += sc[k] * bf2f(vb[(long)k * H_]);
    Xout[(long)b * H_ + h * DK_ + t] = o * inv;
  }
}

// -------------------------- LSTM cell + tag + LN ---------------------------
// One block per batch row b; 512 threads (thread r owns hidden unit r).

__global__ void cell_tag_kernel(const float* __restrict__ gates,
                                float* __restrict__ c,
                                __bf16* __restrict__ hbf,
                                const float* __restrict__ Wtag,
                                const float* __restrict__ btag,
                                const float* __restrict__ lng,
                                const float* __restrict__ lnb,
                                float* __restrict__ out, int t) {
  int b = blockIdx.x;
  int r = threadIdx.x;                         // 0..511
  const float* g = gates + (long)b * (4 * RH_);
  float ig = sigm(g[r]);
  float fg = sigm(g[RH_ + r]);
  float gg = tanhf(g[2 * RH_ + r]);
  float og = sigm(g[3 * RH_ + r]);
  float cc = fg * c[b * RH_ + r] + ig * gg;
  c[b * RH_ + r] = cc;
  float h = og * tanhf(cc);
  hbf[b * RH_ + r] = f2bf(h);

  __shared__ float part[16 * NT_];
  __shared__ float yv[NT_];
  int lane = r & 31, wv = r >> 5;
#pragma unroll
  for (int j = 0; j < NT_; ++j) {
    float p = h * Wtag[j * RH_ + r];
    for (int off = 16; off; off >>= 1) p += __shfl_xor(p, off, 32);
    if (lane == 0) part[wv * NT_ + j] = p;
  }
  __syncthreads();
  if (r < NT_) {
    float y = btag[r];
#pragma unroll
    for (int w = 0; w < 16; ++w) y += part[w * NT_ + r];
    yv[r] = y;
  }
  __syncthreads();
  if (r < NT_) {
    float mu = 0.0f;
#pragma unroll
    for (int j = 0; j < NT_; ++j) mu += yv[j];
    mu *= (1.0f / NT_);
    float var = 0.0f;
#pragma unroll
    for (int j = 0; j < NT_; ++j) { float d = yv[j] - mu; var += d * d; }
    var *= (1.0f / NT_);
    float yn = (yv[r] - mu) * rsqrtf(var + 1e-5f) * lng[r] + lnb[r];
    out[(long)b * NT_ * T_ + r * T_ + t] = yn;   // [B, NT, T]
  }
}

// -------------------------- host side --------------------------------------

static inline void launch_gemm(hipStream_t s, const __bf16* A, int lda,
                               const __bf16* W, int ldw,
                               float* Cf, __bf16* Cbf, int ldc,
                               const float* bias, const float* addsrc,
                               long add_stride, int M, int N, int K) {
  if (M >= 512) {  // big parallel GEMMs: 32x64 wave tiles (A pair shares B)
    long waves = (long)(M / 32) * (N / 64);
    int blocks = (int)((waves + 7) / 8);
    gemm_bf16_kernel<2><<<blocks, 256, 0, s>>>(A, lda, W, ldw, Cf, Cbf, ldc,
                                               bias, addsrc, add_stride, M, N, K);
  } else {         // latency-bound small-M GEMMs: maximize waves in flight
    long waves = (long)(M / 16) * (N / 64);
    int blocks = (int)((waves + 7) / 8);
    gemm_bf16_kernel<1><<<blocks, 256, 0, s>>>(A, lda, W, ldw, Cf, Cbf, ldc,
                                               bias, addsrc, add_stride, M, N, K);
  }
}

static inline void launch_conv(hipStream_t s, const float* in, __bf16* out, long n) {
  f32_to_bf16_kernel<<<(int)((n + 255) / 256), 256, 0, s>>>(in, out, n);
}

extern "C" void kernel_launch(void* const* d_in, const int* in_sizes, int n_in,
                              void* d_out, int out_size, void* d_ws, size_t ws_size,
                              hipStream_t stream) {
  (void)in_sizes; (void)n_in; (void)out_size; (void)ws_size;
  const float* slot = (const float*)d_in[0];
  const float* utt  = (const float*)d_in[1];
  const float* Wq   = (const float*)d_in[2];  const float* bq  = (const float*)d_in[3];
  const float* Wk   = (const float*)d_in[4];  const float* bk  = (const float*)d_in[5];
  const float* Wv   = (const float*)d_in[6];  const float* bv  = (const float*)d_in[7];
  const float* Wo   = (const float*)d_in[8];  const float* bo  = (const float*)d_in[9];
  const float* Win  = (const float*)d_in[10]; const float* bin = (const float*)d_in[11];
  const float* Wih  = (const float*)d_in[12]; const float* Whh = (const float*)d_in[13];
  const float* bih  = (const float*)d_in[14]; const float* bhh = (const float*)d_in[15];
  const float* Wtag = (const float*)d_in[16]; const float* btag= (const float*)d_in[17];
  const float* lng  = (const float*)d_in[18]; const float* lnb = (const float*)d_in[19];
  float* out = (float*)d_out;

  char* ws = (char*)d_ws;
  size_t off = 0;
  auto alloc = [&](size_t bytes) -> char* {
    char* p = ws + off;
    off = (off + bytes + 255) & ~(size_t)255;
    return p;
  };

  __bf16* slot_bf  = (__bf16*)alloc((size_t)B_ * H_ * 2);
  __bf16* utt_bf   = (__bf16*)alloc((size_t)B_ * T_ * H_ * 2);
  __bf16* Wq_bf    = (__bf16*)alloc((size_t)H_ * H_ * 2);
  __bf16* Wk_bf    = (__bf16*)alloc((size_t)H_ * H_ * 2);
  __bf16* Wv_bf    = (__bf16*)alloc((size_t)H_ * H_ * 2);
  __bf16* Wo_bf    = (__bf16*)alloc((size_t)H_ * H_ * 2);
  __bf16* Win_bf   = (__bf16*)alloc((size_t)RIN_ * 2 * H_ * 2);
  __bf16* Wih_bf   = (__bf16*)alloc((size_t)4 * RH_ * RIN_ * 2);
  __bf16* Whh_bf   = (__bf16*)alloc((size_t)4 * RH_ * RH_ * 2);
  float*  Qf       = (float*) alloc((size_t)B_ * H_ * 4);
  __bf16* K_bf     = (__bf16*)alloc((size_t)B_ * T_ * H_ * 2);
  __bf16* V_bf     = (__bf16*)alloc((size_t)B_ * T_ * H_ * 2);
  float*  Xattn_f  = (float*) alloc((size_t)B_ * H_ * 4);
  __bf16* Xattn_bf = (__bf16*)alloc((size_t)B_ * H_ * 2);
  float*  Xctx_f   = (float*) alloc((size_t)B_ * H_ * 4);
  __bf16* Xctx_bf  = (__bf16*)alloc((size_t)B_ * H_ * 2);
  float*  Actx_f   = (float*) alloc((size_t)B_ * RIN_ * 4);
  float*  prex_f   = (float*) alloc((size_t)B_ * T_ * RIN_ * 4);
  __bf16* lstmin_bf= (__bf16*)alloc((size_t)B_ * T_ * RIN_ * 2);
  float*  gatesin_f= (float*) alloc((size_t)B_ * T_ * 4 * RH_ * 4);  // 134 MB
  float*  gstep_f  = (float*) alloc((size_t)B_ * 4 * RH_ * 4);
  float*  c_f      = (float*) alloc((size_t)B_ * RH_ * 4);
  __bf16* h_bf     = (__bf16*)alloc((size_t)B_ * RH_ * 2);
  float*  bcomb    = (float*) alloc((size_t)4 * RH_ * 4);

  // 1) convert inputs & weights to bf16
  launch_conv(stream, slot, slot_bf, (long)B_ * H_);
  launch_conv(stream, utt,  utt_bf,  (long)B_ * T_ * H_);
  launch_conv(stream, Wq,   Wq_bf,   (long)H_ * H_);
  launch_conv(stream, Wk,   Wk_bf,   (long)H_ * H_);
  launch_conv(stream, Wv,   Wv_bf,   (long)H_ * H_);
  launch_conv(stream, Wo,   Wo_bf,   (long)H_ * H_);
  launch_conv(stream, Win,  Win_bf,  (long)RIN_ * 2 * H_);
  launch_conv(stream, Wih,  Wih_bf,  (long)4 * RH_ * RIN_);
  launch_conv(stream, Whh,  Whh_bf,  (long)4 * RH_ * RH_);
  bias_combine_kernel<<<(4 * RH_ + 255) / 256, 256, 0, stream>>>(bih, bhh, bcomb, 4 * RH_);

  // 2) projections (WMMA GEMMs)
  launch_gemm(stream, slot_bf, H_, Wq_bf, H_, Qf, nullptr, H_, bq,
              nullptr, 0, B_, H_, H_);
  launch_gemm(stream, utt_bf, H_, Wk_bf, H_, nullptr, K_bf, H_, bk,
              nullptr, 0, B_ * T_, H_, H_);
  launch_gemm(stream, utt_bf, H_, Wv_bf, H_, nullptr, V_bf, H_, bv,
              nullptr, 0, B_ * T_, H_, H_);

  // 3) attention -> Xattn, then output projection -> Xctx
  attn_kernel<<<B_ * HEADS_, T_, 0, stream>>>(Qf, K_bf, V_bf, Xattn_f);
  launch_conv(stream, Xattn_f, Xattn_bf, (long)B_ * H_);
  launch_gemm(stream, Xattn_bf, H_, Wo_bf, H_, Xctx_f, nullptr, H_, bo,
              nullptr, 0, B_, H_, H_);
  launch_conv(stream, Xctx_f, Xctx_bf, (long)B_ * H_);

  // 4) input-side LSTM precompute:
  //    A_ctx = Xctx @ W_in[:,H:]^T + b_in ;  pre_x = utt @ W_in[:,:H]^T
  launch_gemm(stream, Xctx_bf, H_, Win_bf + H_, 2 * H_, Actx_f, nullptr, RIN_,
              bin, nullptr, 0, B_, RIN_, H_);
  launch_gemm(stream, utt_bf, H_, Win_bf, 2 * H_, prex_f, nullptr, RIN_,
              nullptr, nullptr, 0, B_ * T_, RIN_, H_);
  {
    long n = (long)B_ * T_ * RIN_;
    lstm_in_kernel<<<(int)((n + 255) / 256), 256, 0, stream>>>(prex_f, Actx_f, lstmin_bf);
  }
  //    gates_in = sigmoid(...) @ W_ih^T + (b_ih + b_hh)   for all (b,t)
  launch_gemm(stream, lstmin_bf, RIN_, Wih_bf, RIN_, gatesin_f, nullptr, 4 * RH_,
              bcomb, nullptr, 0, B_ * T_, 4 * RH_, RIN_);

  // 5) recurrence: only h @ W_hh^T + cell remain sequential
  init_state_kernel<<<(B_ * RH_ + 255) / 256, 256, 0, stream>>>(c_f, h_bf, B_ * RH_);
  for (int t = 0; t < T_; ++t) {
    launch_gemm(stream, h_bf, RH_, Whh_bf, RH_, gstep_f, nullptr, 4 * RH_,
                nullptr, gatesin_f + (size_t)t * 4 * RH_, (long)T_ * 4 * RH_,
                B_, 4 * RH_, RH_);
    cell_tag_kernel<<<B_, RH_, 0, stream>>>(gstep_f, c_f, h_bf,
                                            Wtag, btag, lng, lnb, out, t);
  }
}
